// MovingWindowHyperbolicForecaster_23871428232052
// MI455X (gfx1250) — matrix operand order repaired and checked by reading, not compiled, not run on hardware
//
#include <hip/hip_runtime.h>

// ---------------- problem constants ----------------
#define BDIM 256
constexpr int Bn   = 128;
constexpr int Ln   = 720;
constexpr int Fn   = 128;
constexpr int SEGc = 24;
constexpr int WINc = 15;
constexpr int NPRED = 4;
constexpr int Dd   = 128;
constexpr int Hh   = 512;
constexpr float DECAYc = 0.9f;
constexpr float EPSf   = 1e-6f;
constexpr float MAXNf  = 1.0f - 1e-5f;

// workspace layout (bytes)
constexpr size_t OFF_WCAT = 0;            // [96][128]  f16  = 24576 B
constexpr size_t OFF_W1H  = 24576;        // [128][512] f16  = 131072 B
constexpr size_t OFF_W2H  = 155648;       // [512][32]  f16  = 32768 B (cols 24..31 zero)
constexpr size_t OFF_BCAT = 188416;       // [128]      f32

// dynamic LDS layout (bytes)
constexpr int LDS_ZBUF  = 15 * 16 * 128 * 4;   // 122880  z ring [slot][row][128] f32
constexpr int LDS_TOTAL = LDS_ZBUF + 33664;    // 156544

typedef __attribute__((ext_vector_type(16))) _Float16 v16h;
typedef __attribute__((ext_vector_type(8)))  _Float16 v8h;
typedef __attribute__((ext_vector_type(8)))  float    v8f;

__device__ __forceinline__ v16h cat16(v8h lo, v8h hi) {
  return __builtin_shufflevector(lo, hi, 0,1,2,3,4,5,6,7,8,9,10,11,12,13,14,15);
}
__device__ __forceinline__ float artanh_pos(float x) {
  x = fminf(x, 1.0f - 1e-7f);
  return 0.5f * logf((1.0f + x) / (1.0f - x));
}

// ---------------- weight pack: f32 -> f16 (concat / pad) ----------------
__global__ void pack_weights(const float* __restrict__ Wt, const float* __restrict__ Wc,
                             const float* __restrict__ Wf, const float* __restrict__ Wr,
                             const float* __restrict__ bt, const float* __restrict__ bc_,
                             const float* __restrict__ bf_, const float* __restrict__ br_,
                             const float* __restrict__ W1, const float* __restrict__ W2,
                             char* __restrict__ ws)
{
  _Float16* wcat = (_Float16*)(ws + OFF_WCAT);
  _Float16* w1h  = (_Float16*)(ws + OFF_W1H);
  _Float16* w2h  = (_Float16*)(ws + OFF_W2H);
  float*    bcat = (float*)   (ws + OFF_BCAT);
  int idx = blockIdx.x * BDIM + threadIdx.x;
  if (idx < 12288) {                       // Wcat[k=i*24+s][d]
    int k = idx >> 7, d = idx & 127;
    int i = k / 24, s = k % 24;
    const float* Wi = (i == 0) ? Wt : (i == 1) ? Wc : (i == 2) ? Wf : Wr;
    wcat[idx] = (_Float16)Wi[s * 128 + d];
  } else if (idx < 12288 + 65536) {        // W1 straight copy
    int j = idx - 12288;
    w1h[j] = (_Float16)W1[j];
  } else if (idx < 12288 + 65536 + 16384) {// W2 padded to 32 cols
    int j = idx - (12288 + 65536);
    int k = j >> 5, n = j & 31;
    w2h[j] = (n < SEGc) ? (_Float16)W2[k * SEGc + n] : (_Float16)0.0f;
  } else if (idx < 12288 + 65536 + 16384 + 128) {
    int d = idx - (12288 + 65536 + 16384);
    bcat[d] = bt[d] + bc_[d] + bf_[d] + br_[d];
  }
}

// ---------------- fused forecaster: 1 WG = 16 collapsed rows ----------------
__global__ __launch_bounds__(BDIM, 1)
void hyper_forecast(const float* __restrict__ trend, const float* __restrict__ sea_c,
                    const float* __restrict__ sea_f, const float* __restrict__ resid,
                    const _Float16* __restrict__ wcat, const float* __restrict__ bcat,
                    const _Float16* __restrict__ w1h,  const float* __restrict__ b1,
                    const _Float16* __restrict__ w2h,  const float* __restrict__ b2,
                    const float* __restrict__ alpha_p, float* __restrict__ out)
{
  extern __shared__ char smem[];
  float*    zbuf = (float*)smem;                    // [15][16][128] f32
  char*     R    = smem + LDS_ZBUF;
  _Float16* t0h  = (_Float16*)(R);                  // [16][128] f16
  _Float16* hld  = (_Float16*)(R + 4096);           // [16][512] f16
  float*    vavg = (float*)(R + 20480);             // [16][128] f32
  float*    sqld = (float*)(R + 28672);             // [16][15]  f32
  float*    pxld = (float*)(R + 29696);             // [16][14]  f32
  float*    pyld = (float*)(R + 30592);             // [16][14]  f32
  float*    rowA = (float*)(R + 31488);             // [16]
  float*    rowB = (float*)(R + 31552);             // [16]
  float*    oacc = (float*)(R + 31616);             // [16][32]  f32

  const int tid  = threadIdx.x;
  const int wave = tid >> 5, lane = tid & 31;
  const int wg   = blockIdx.x;
  const int b    = wg >> 3;
  const int f0   = (wg & 7) << 4;
  const float alpha = alpha_p[0];

  // ===== Stage A: tan = x_cat @ Wcat + bcat for the 15 window segments =====
  // A tile = 16 rows (f0..f0+15) of one segment, K=96, N=128 (8 n-tiles).
  for (int seg = wave; seg < WINc; seg += 8) {
    const int labs = (WINc + seg) * SEGc;           // absolute time of segment start
    v16h afr[3];
    #pragma unroll
    for (int ks = 0; ks < 3; ++ks) {
      _Float16 tmp[16];
      #pragma unroll
      for (int half = 0; half < 2; ++half) {
        int c = ks * 32 + ((lane & 16) ? 8 : 0) + half * 16;  // chunk of 8 K-values
        int i = c / SEGc, sb = c % SEGc;                      // never crosses input boundary
        const float* src = (i == 0) ? trend : (i == 1) ? sea_c : (i == 2) ? sea_f : resid;
        const float* p = src + ((size_t)(b * Ln + labs + sb)) * Fn + f0 + (lane & 15);
        #pragma unroll
        for (int t = 0; t < 8; ++t) tmp[half * 8 + t] = (_Float16)p[t * Fn];
      }
      v16h a;
      #pragma unroll
      for (int e = 0; e < 16; ++e) a[e] = tmp[e];
      afr[ks] = a;
    }
    for (int nt = 0; nt < 8; ++nt) {
      v8f acc = {};
      #pragma unroll
      for (int ks = 0; ks < 3; ++ks) {
        const _Float16* wp = wcat + (ks * 32 + ((lane & 16) ? 8 : 0)) * Dd + nt * 16 + (lane & 15);
        v16h bfr;
        #pragma unroll
        for (int e = 0; e < 16; ++e) bfr[e] = wp[((e & 7) + ((e >> 3) << 4)) * Dd];
        acc = __builtin_amdgcn_wmma_f32_16x16x32_f16(false, afr[ks], false, bfr,
                                                     (short)0, acc, false, false);
      }
      int   nidx = nt * 16 + (lane & 15);
      float bias = bcat[nidx];
      int   mhi  = (lane & 16) ? 8 : 0;
      #pragma unroll
      for (int j = 0; j < 8; ++j)
        zbuf[(seg * 16 + mhi + j) * Dd + nidx] = acc[j] + bias;
    }
  }
  __syncthreads();

  // ===== expmap0 on all 240 window vectors =====
  if (tid < WINc * 16) {
    float* p = zbuf + tid * Dd;
    float sq = 0.f;
    for (int k = 0; k < Dd; ++k) { float v = p[k]; sq += v * v; }
    float n  = sqrtf(fmaxf(sq, EPSf * EPSf));
    float th = tanhf(n);
    float sc = th / n;
    if (th > MAXNf) sc *= MAXNf / th;                // _project
    for (int k = 0; k < Dd; ++k) p[k] *= sc;
  }
  __syncthreads();

  int head = 0;                                      // ring: logical j -> phys (head+j)%15
  for (int it = 0; it < NPRED; ++it) {
    // --- 1a: |z|^2 per (row, slot) ---
    if (tid < WINc * 16) {
      int slot = tid >> 4, row = tid & 15;
      const float* p = zbuf + (slot * 16 + row) * Dd;
      float sq = 0.f;
      for (int k = 0; k < Dd; ++k) sq += p[k] * p[k];
      sqld[row * WINc + slot] = sq;
    }
    __syncthreads();

    // --- 1b: per-pair logmap reduced to scalar coefficients ---
    // vel_j = S_j * (-A_j * x_j + B_j * y_j);  avg_v = sum_j w_j vel_j
    if (tid < 16 * (WINc - 1)) {
      int row = tid / (WINc - 1), j = tid % (WINc - 1);
      int px = (head + j) % WINc, py = (head + j + 1) % WINc;
      const float* xp = zbuf + (px * 16 + row) * Dd;
      const float* yp = zbuf + (py * 16 + row) * Dd;
      float xy = 0.f;
      for (int k = 0; k < Dd; ++k) xy += xp[k] * yp[k];
      float xx = sqld[row * WINc + px], yy = sqld[row * WINc + py];
      float A   = 1.f - 2.f * xy + yy;               // coef on (-x) in mobius_add(-x,y)
      float Bc  = 1.f - xx;                          // coef on y
      float den = fmaxf(1.f - 2.f * xy + xx * yy, EPSf);
      float uu  = A * A * xx - 2.f * A * Bc * xy + Bc * Bc * yy;   // |num|^2
      float dd  = fmaxf(uu / (den * den), EPSf * EPSf);
      float nd  = sqrtf(dd);
      float fac = fmaxf(1.f - xx, EPSf);             // 2/(sqc*lam)
      float S   = fac * artanh_pos(nd) / (den * nd);
      // decay weights w_j = DECAY^(13-j)/sum
      float sum = 0.f, pw = 1.f;
      for (int e = 0; e < WINc - 1; ++e) { sum += pw; pw *= DECAYc; }
      float wj = 1.f;
      for (int e = 0; e < (WINc - 2 - j); ++e) wj *= DECAYc;
      wj /= sum;
      pxld[row * (WINc - 1) + j] = -wj * S * A;
      pyld[row * (WINc - 1) + j] =  wj * S * Bc;
    }
    __syncthreads();

    // --- 2: v = alpha * sum_j g_j * z_j  (race-free linear combination) ---
    for (int idx = tid; idx < 16 * Dd; idx += BDIM) {
      int row = idx >> 7, k = idx & 127;
      float a = 0.f;
      for (int j = 0; j < WINc; ++j) {
        float g = 0.f;
        if (j < WINc - 1) g += pxld[row * (WINc - 1) + j];
        if (j > 0)        g += pyld[row * (WINc - 1) + j - 1];
        int ph = (head + j) % WINc;
        a += g * zbuf[(ph * 16 + row) * Dd + k];
      }
      vavg[idx] = alpha * a;
    }
    __syncthreads();

    // --- 3: expmap(z_last, v) scalars per row ---
    if (tid < 16) {
      int row = tid;
      int pl = (head + WINc - 1) % WINc;
      const float* x = zbuf + (pl * 16 + row) * Dd;
      const float* v = vavg + row * Dd;
      float xv = 0.f, vv = 0.f;
      for (int k = 0; k < Dd; ++k) { xv += x[k] * v[k]; vv += v[k] * v[k]; }
      float xx  = sqld[row * WINc + pl];
      float n   = sqrtf(fmaxf(vv, EPSf * EPSf));
      float tn  = tanhf(n / fmaxf(1.f - xx, EPSf));  // tanh(lam*n/2)
      float tc  = tn / n;                            // second = tc * v
      float xy2 = tc * xv;
      float y2  = tc * tc * vv;
      float a   = 1.f + 2.f * xy2 + y2;
      float bb  = 1.f - xx;
      float den = fmaxf(1.f + 2.f * xy2 + xx * y2, EPSf);
      rowA[row] = a / den;
      rowB[row] = bb * tc / den;
    }
    __syncthreads();

    // --- 4: z_next -> evicted ring slot `head` ---
    {
      int pl = (head + WINc - 1) % WINc;
      for (int idx = tid; idx < 16 * Dd; idx += BDIM) {
        int row = idx >> 7, k = idx & 127;
        zbuf[(head * 16 + row) * Dd + k] =
            rowA[row] * zbuf[(pl * 16 + row) * Dd + k] + rowB[row] * vavg[idx];
      }
    }
    __syncthreads();

    // --- 5: project(z_next) + t0 = logmap0(z_next) as f16 A-tile ---
    if (tid < 16) {
      int row = tid;
      float* p = zbuf + (head * 16 + row) * Dd;
      float sq = 0.f;
      for (int k = 0; k < Dd; ++k) sq += p[k] * p[k];
      float n  = sqrtf(fmaxf(sq, EPSf * EPSf));
      float sc = (n > MAXNf) ? (MAXNf / n) : 1.f;
      float np = n * sc;
      float l0 = artanh_pos(np) / np;
      for (int k = 0; k < Dd; ++k) {
        float z = p[k] * sc;
        p[k] = z;
        t0h[row * Dd + k] = (_Float16)(l0 * z);
      }
    }
    __syncthreads();

    // --- 6: GEMM1  h = relu(t0 @ W1 + b1), 16x128 @ 128x512 ---
    for (int q = 0; q < 4; ++q) {
      int n0 = (wave * 4 + q) * 16;
      v8f acc = {};
      #pragma unroll
      for (int ks = 0; ks < 4; ++ks) {
        int kb = ks * 32 + ((lane & 16) ? 8 : 0);
        int m  = lane & 15;
        v8h alo = *(const v8h*)(t0h + m * Dd + kb);
        v8h ahi = *(const v8h*)(t0h + m * Dd + kb + 16);
        const _Float16* wp = w1h + ((size_t)kb) * Hh + n0 + (lane & 15);
        __builtin_prefetch(wp, 0, 1);
        v16h bfr;
        #pragma unroll
        for (int e = 0; e < 16; ++e) bfr[e] = wp[(size_t)((e & 7) + ((e >> 3) << 4)) * Hh];
        acc = __builtin_amdgcn_wmma_f32_16x16x32_f16(false, cat16(alo, ahi), false, bfr,
                                                     (short)0, acc, false, false);
      }
      int   nidx = n0 + (lane & 15);
      float bias = b1[nidx];
      int   mhi  = (lane & 16) ? 8 : 0;
      #pragma unroll
      for (int j = 0; j < 8; ++j) {
        float hv = fmaxf(acc[j] + bias, 0.f);
        hld[(mhi + j) * Hh + nidx] = (_Float16)hv;
      }
    }
    // init output accumulator with b2 (cols >=24 zero-padded)
    for (int idx = tid; idx < 16 * 32; idx += BDIM) {
      int n = idx & 31;
      oacc[idx] = (n < SEGc) ? b2[n] : 0.f;
    }
    __syncthreads();

    // --- 7: GEMM2  seg = h @ W2, 16x512 @ 512x32; K split over 4 waves/tile ---
    {
      int nt = wave & 1, kq = wave >> 1;
      int n0 = nt * 16;
      v8f acc = {};
      #pragma unroll
      for (int ks4 = 0; ks4 < 4; ++ks4) {
        int ks = kq * 4 + ks4;
        int kb = ks * 32 + ((lane & 16) ? 8 : 0);
        int m  = lane & 15;
        v8h alo = *(const v8h*)(hld + m * Hh + kb);
        v8h ahi = *(const v8h*)(hld + m * Hh + kb + 16);
        const _Float16* wp = w2h + kb * 32 + n0 + (lane & 15);
        v16h bfr;
        #pragma unroll
        for (int e = 0; e < 16; ++e) bfr[e] = wp[((e & 7) + ((e >> 3) << 4)) * 32];
        acc = __builtin_amdgcn_wmma_f32_16x16x32_f16(false, cat16(alo, ahi), false, bfr,
                                                     (short)0, acc, false, false);
      }
      int nidx = n0 + (lane & 15);
      int mhi  = (lane & 16) ? 8 : 0;
      #pragma unroll
      for (int j = 0; j < 8; ++j)
        atomicAdd(&oacc[(mhi + j) * 32 + nidx], acc[j]);   // ds_add_f32
    }
    __syncthreads();

    // --- 8: scatter predictions (B, 96, F) ---
    if (tid < 16 * SEGc) {
      int row = tid & 15, s = tid >> 4;
      out[((size_t)b * (NPRED * SEGc) + it * SEGc + s) * Fn + f0 + row] = oacc[row * 32 + s];
    }
    __syncthreads();
    head = (head + 1) % WINc;
  }
}

// ---------------- host entry ----------------
extern "C" void kernel_launch(void* const* d_in, const int* in_sizes, int n_in,
                              void* d_out, int out_size, void* d_ws, size_t ws_size,
                              hipStream_t stream)
{
  const float* trend = (const float*)d_in[0];
  const float* sc    = (const float*)d_in[1];
  const float* sf    = (const float*)d_in[2];
  const float* rs    = (const float*)d_in[3];
  const float* Wt    = (const float*)d_in[4];
  const float* bt    = (const float*)d_in[5];
  const float* Wc    = (const float*)d_in[6];
  const float* bc    = (const float*)d_in[7];
  const float* Wf    = (const float*)d_in[8];
  const float* bf    = (const float*)d_in[9];
  const float* Wr    = (const float*)d_in[10];
  const float* br    = (const float*)d_in[11];
  const float* alpha = (const float*)d_in[12];
  const float* W1    = (const float*)d_in[13];
  const float* b1    = (const float*)d_in[14];
  const float* W2    = (const float*)d_in[15];
  const float* b2    = (const float*)d_in[16];
  char* ws = (char*)d_ws;

  pack_weights<<<(12288 + 65536 + 16384 + 128 + BDIM - 1) / BDIM, BDIM, 0, stream>>>(
      Wt, Wc, Wf, Wr, bt, bc, bf, br, W1, W2, ws);

  const _Float16* wcat = (const _Float16*)(ws + OFF_WCAT);
  const _Float16* w1h  = (const _Float16*)(ws + OFF_W1H);
  const _Float16* w2h  = (const _Float16*)(ws + OFF_W2H);
  const float*    bcat = (const float*)(ws + OFF_BCAT);

  hyper_forecast<<<1024, BDIM, LDS_TOTAL, stream>>>(
      trend, sc, sf, rs, wcat, bcat, w1h, b1, w2h, b2, alpha, (float*)d_out);
}
